// Model_41618233098842
// MI455X (gfx1250) — compile-verified, matrix operand order checked by reading, not run on hardware
//
#include <hip/hip_runtime.h>
#include <math.h>

typedef __attribute__((ext_vector_type(16))) _Float16 v16h;
typedef __attribute__((ext_vector_type(8)))  _Float16 v8h;
typedef __attribute__((ext_vector_type(8)))  float    v8f;

#define DEV __device__ __forceinline__

DEV v8f wmma_f16(v16h a, v16h b, v8f c) {
  // D = A(16x32 f16) x B(32x16 f16) + C(16x16 f32)
  return __builtin_amdgcn_wmma_f32_16x16x32_f16(false, a, false, b,
                                                (short)0, c, false, false);
}

// Build a 16x32 f16 fragment slice from an LDS f16 row (row-major, row holds >=32
// contiguous K values starting at `base`).  Per documented layout: lane < 16 uses
// K = {kb..kb+7, kb+16..kb+23} with kb=0, lanes 16..31 use kb=8.
DEV v16h load_frag(const _Float16* base, int kb) {
  v8h lo = *(const v8h*)(base + kb);
  v8h hi = *(const v8h*)(base + kb + 16);
  v16h r;
#pragma unroll
  for (int i = 0; i < 8; ++i) { r[i] = lo[i]; r[8 + i] = hi[i]; }
  return r;
}

// Same, but source row is f32 (converted on the fly).
DEV v16h frag_from_f32(const float* base, int kb) {
  v16h r;
#pragma unroll
  for (int i = 0; i < 8; ++i) {
    r[i]     = (_Float16)base[kb + i];
    r[8 + i] = (_Float16)base[kb + 16 + i];
  }
  return r;
}

DEV float sigmoidf_(float x) { return 1.f / (1.f + __expf(-x)); }

// ---------------------------------------------------------------------------
// Generic mixed-precision GEMM:  C[M,N] = A[M,K] * B + bias + res
//   b_layout==0 : B is W[N,K] row-major (i.e. compute A @ W^T), ldb = ldw
//   b_layout==1 : B is [K,N] row-major (activation matrix),     ldb = N stride
// Block: 128 threads (4 waves), tile 64(M) x 64(N), K stepped by 32.
// ---------------------------------------------------------------------------
__global__ __launch_bounds__(128)
void gemm_wmma(const float* __restrict__ A, const float* __restrict__ B,
               const float* __restrict__ bias, const float* __restrict__ res,
               float* __restrict__ C,
               int K, int lda, int ldb, int ldc, int b_layout, int Nreal,
               long long strideA, long long strideB, long long strideC) {
  __shared__ _Float16 As[64 * 32];
  __shared__ _Float16 Bs[64 * 32];
  const int tid  = threadIdx.x;
  const int wave = tid >> 5;
  const int lane = tid & 31;
  const int m0 = blockIdx.x * 64;
  const int n0 = blockIdx.y * 64;
  const long long zb = blockIdx.z;
  A += zb * strideA;
  B += zb * strideB;
  C += zb * strideC;
  if (res) res += zb * strideC;

  v8f acc[4];
#pragma unroll
  for (int i = 0; i < 4; ++i)
#pragma unroll
    for (int j = 0; j < 8; ++j) acc[i][j] = 0.f;

  const int kb = (lane < 16) ? 0 : 8;
  const int lr = lane & 15;

  for (int k0 = 0; k0 < K; k0 += 32) {
    // stage A tile [64 m][32 k], coalesced along k
    for (int idx = tid; idx < 64 * 32; idx += 128) {
      int r = idx >> 5, c = idx & 31;
      As[r * 32 + c] = (_Float16)A[(long long)(m0 + r) * lda + (k0 + c)];
    }
    if (k0 + 32 < K)
      __builtin_prefetch(&A[(long long)m0 * lda + (k0 + 32)], 0, 1);
    // stage B tile into Bs[n][k]
    if (b_layout == 0) {
      for (int idx = tid; idx < 64 * 32; idx += 128) {
        int n = idx >> 5, c = idx & 31;
        float v = (n0 + n < Nreal) ? B[(long long)(n0 + n) * ldb + (k0 + c)] : 0.f;
        Bs[n * 32 + c] = (_Float16)v;
      }
    } else {
      for (int idx = tid; idx < 64 * 32; idx += 128) {
        int kk = idx >> 6, n = idx & 63;
        float v = (n0 + n < Nreal) ? B[(long long)(k0 + kk) * ldb + (n0 + n)] : 0.f;
        Bs[n * 32 + kk] = (_Float16)v;
      }
    }
    __syncthreads();
    v16h af = load_frag(&As[(wave * 16 + lr) * 32], kb);
#pragma unroll
    for (int nt = 0; nt < 4; ++nt) {
      v16h bf = load_frag(&Bs[(nt * 16 + lr) * 32], kb);
      acc[nt] = wmma_f16(af, bf, acc[nt]);
    }
    __syncthreads();
  }

  // C layout: lanes 0-15 -> M = r, lanes 16-31 -> M = r + 8
  const int mh = (lane >= 16) ? 8 : 0;
#pragma unroll
  for (int nt = 0; nt < 4; ++nt) {
    int gn = n0 + nt * 16 + lr;
    if (gn >= Nreal) continue;
    float bv = bias ? bias[gn] : 0.f;
#pragma unroll
    for (int r = 0; r < 8; ++r) {
      long long gm = m0 + wave * 16 + r + mh;
      float v = acc[nt][r] + bv;
      if (res) v += res[gm * ldc + gn];
      C[gm * ldc + gn] = v;
    }
  }
}

// ---------------------------------------------------------------------------
// GRU: one wave handles 16 sequences; per step gh = h @ Whh^T + bhh via WMMA
// (12 N-tiles x 2 K-chunks = 24 WMMAs), then fused gate math.
// din == 1, E == 64, 3E == 192.
// ---------------------------------------------------------------------------
__global__ __launch_bounds__(32)
void gru_kernel(const float* __restrict__ x,   // [4096,36]
                const float* __restrict__ Wih, // [192]
                const float* __restrict__ Whh, // [192,64]
                const float* __restrict__ bih, const float* __restrict__ bhh,
                float* __restrict__ out,       // [4096,36,64]
                int t0, int Tlen) {
  __shared__ _Float16 Wh[192 * 64];
  __shared__ float hbuf[16 * 64];
  __shared__ float xt[16 * 16];
  __shared__ float wih_s[192], bih_s[192], bhh_s[192];
  const int lane = threadIdx.x;
  const long long seq0 = (long long)blockIdx.x * 16;

  for (int i = lane; i < 192 * 64; i += 32) Wh[i] = (_Float16)Whh[i];
  for (int i = lane; i < 192; i += 32) {
    wih_s[i] = Wih[i]; bih_s[i] = bih[i]; bhh_s[i] = bhh[i];
  }
  for (int i = lane; i < 16 * 64; i += 32) hbuf[i] = 0.f;
  for (int i = lane; i < 16 * Tlen; i += 32) {
    int m = i / Tlen, t = i % Tlen;
    xt[m * 16 + t] = x[(seq0 + m) * 36 + t0 + t];
  }
  __syncthreads();

  const int kb = (lane < 16) ? 0 : 8;
  const int lr = lane & 15;
  const int mh = (lane >= 16) ? 8 : 0;

  for (int t = 0; t < Tlen; ++t) {
    v16h a0 = frag_from_f32(&hbuf[lr * 64 + 0], kb);
    v16h a1 = frag_from_f32(&hbuf[lr * 64 + 32], kb);
    v8f accs[12];
#pragma unroll
    for (int nt = 0; nt < 12; ++nt) {
      float bz = bhh_s[nt * 16 + lr];
      v8f c;
#pragma unroll
      for (int j = 0; j < 8; ++j) c[j] = bz;
      v16h b0 = load_frag(&Wh[(nt * 16 + lr) * 64 + 0], kb);
      v16h b1 = load_frag(&Wh[(nt * 16 + lr) * 64 + 32], kb);
      c = wmma_f16(a0, b0, c);
      c = wmma_f16(a1, b1, c);
      accs[nt] = c;
    }
    // gates: tiles 0-3 = r, 4-7 = z, 8-11 = n
#pragma unroll
    for (int gt = 0; gt < 4; ++gt) {
      int j = gt * 16 + lr;
      float wr = wih_s[j], wz = wih_s[64 + j], wn = wih_s[128 + j];
      float br = bih_s[j], bz = bih_s[64 + j], bn = bih_s[128 + j];
#pragma unroll
      for (int r = 0; r < 8; ++r) {
        int m = r + mh;
        float xv = xt[m * 16 + t];
        float rr = sigmoidf_(xv * wr + br + accs[gt][r]);
        float zz = sigmoidf_(xv * wz + bz + accs[gt + 4][r]);
        float nn = tanhf(xv * wn + bn + rr * accs[gt + 8][r]);
        float hold = hbuf[m * 64 + j];
        float hnew = (1.f - zz) * nn + zz * hold;
        hbuf[m * 64 + j] = hnew;
        out[((seq0 + m) * 36 + t0 + t) * 64 + j] = hnew;
      }
    }
    __syncthreads();
  }
}

// adj = softmax(relu(nv1 @ nv2), axis=1); store TRANSPOSED: adjT[m*512+n]
__global__ __launch_bounds__(256)
void adj_kernel(const float* __restrict__ nv1, const float* __restrict__ nv2,
                float* __restrict__ adjT) {
  __shared__ float row[512];
  __shared__ float red[256];
  const int n = blockIdx.x, tid = threadIdx.x;
  float v1[10];
#pragma unroll
  for (int k = 0; k < 10; ++k) v1[k] = nv1[n * 10 + k];
  float mx = -1e30f;
  for (int m = tid; m < 512; m += 256) {
    float s = 0.f;
#pragma unroll
    for (int k = 0; k < 10; ++k) s += v1[k] * nv2[k * 512 + m];
    s = fmaxf(s, 0.f);
    row[m] = s;
    mx = fmaxf(mx, s);
  }
  red[tid] = mx; __syncthreads();
  for (int off = 128; off > 0; off >>= 1) {
    if (tid < off) red[tid] = fmaxf(red[tid], red[tid + off]);
    __syncthreads();
  }
  mx = red[0]; __syncthreads();
  float sm = 0.f;
  for (int m = tid; m < 512; m += 256) {
    float e = __expf(row[m] - mx);
    row[m] = e; sm += e;
  }
  red[tid] = sm; __syncthreads();
  for (int off = 128; off > 0; off >>= 1) {
    if (tid < off) red[tid] += red[tid + off];
    __syncthreads();
  }
  float inv = 1.f / red[0];
  for (int m = tid; m < 512; m += 256) adjT[m * 512 + n] = row[m] * inv;
}

// cat[r, 0:64]=h, [64:128]=x1, [128:192]=x2   (all viewed as [R,64])
__global__ void concat_kernel(const float* __restrict__ h,
                              const float* __restrict__ x1,
                              const float* __restrict__ x2,
                              float* __restrict__ cat, long long total) {
  long long idx = (long long)blockIdx.x * 256 + threadIdx.x;
  if (idx >= total) return;
  long long r = idx / 192;
  int c = (int)(idx % 192);
  const float* src = (c < 64) ? h : (c < 128) ? x1 : x2;
  cat[idx] = src[r * 64 + (c & 63)];
}

// LayerNorm over last dim (64): out = LN(in1 [+ in2(flipped t)] [+ res])*g + b
__global__ __launch_bounds__(256)
void ln_kernel(const float* __restrict__ in1, const float* __restrict__ in2,
               const float* __restrict__ res, const float* __restrict__ g,
               const float* __restrict__ b, float* __restrict__ out,
               int rows, int flip2) {
  int row = blockIdx.x * 8 + (threadIdx.x >> 5);
  int lane = threadIdx.x & 31;
  if (row >= rows) return;
  long long r2 = row;
  if (flip2) { int t = row % 36; r2 = (long long)row - t + (35 - t); }
  float v[2];
#pragma unroll
  for (int i = 0; i < 2; ++i) {
    int e = lane + i * 32;
    float xv = in1[(long long)row * 64 + e];
    if (in2) xv += in2[r2 * 64 + e];
    if (res) xv += res[(long long)row * 64 + e];
    v[i] = xv;
  }
  float s = v[0] + v[1], sq = v[0] * v[0] + v[1] * v[1];
#pragma unroll
  for (int off = 16; off > 0; off >>= 1) {
    s  += __shfl_xor(s, off, 32);
    sq += __shfl_xor(sq, off, 32);
  }
  float mu = s * (1.f / 64.f);
  float var = sq * (1.f / 64.f) - mu * mu;
  float rs = rsqrtf(var + 1e-5f);
#pragma unroll
  for (int i = 0; i < 2; ++i) {
    int e = lane + i * 32;
    out[(long long)row * 64 + e] = (v[i] - mu) * rs * g[e] + b[e];
  }
}

// time-flip a [R,64] buffer (T=36 within each sequence)
__global__ void flip_kernel(const float* __restrict__ in, float* __restrict__ out,
                            long long total) {
  long long idx = (long long)blockIdx.x * 256 + threadIdx.x;
  if (idx >= total) return;
  long long row = idx >> 6;
  int e = (int)(idx & 63);
  int t = (int)(row % 36);
  long long r2 = row - t + (35 - t);
  out[idx] = in[r2 * 64 + e];
}

// depthwise causal conv (K=4) + SiLU: reads xc = xz[:, :128], writes xa[R,128]
__global__ void dwconv_silu_kernel(const float* __restrict__ xz,
                                   const float* __restrict__ w,
                                   const float* __restrict__ b,
                                   float* __restrict__ xa, long long total) {
  long long idx = (long long)blockIdx.x * 256 + threadIdx.x;
  if (idx >= total) return;
  int c = (int)(idx & 127);
  long long row = idx >> 7;
  long long m = row / 36;
  int t = (int)(row % 36);
  float s = b[c];
#pragma unroll
  for (int k = 0; k < 4; ++k) {
    int ts = t - 3 + k;
    if (ts >= 0) s += xz[(m * 36 + ts) * 256 + c] * w[c * 4 + k];
  }
  xa[idx] = s * sigmoidf_(s);
}

// Selective scan: block = one sequence, thread = one channel (di=128, ds=16).
// Fuses softplus(dt@dt_w^T+dt_b), recurrence, +u*D and SiLU(z) gating.
__global__ __launch_bounds__(128)
void scan_kernel(const float* __restrict__ xdbl, // [R,48] (dt 0:4, B 4:20, C 20:36)
                 const float* __restrict__ xa,   // [R,128]
                 const float* __restrict__ xz,   // [R,256] (z = cols 128:256)
                 const float* __restrict__ A_log, const float* __restrict__ D,
                 const float* __restrict__ dt_w, const float* __restrict__ dt_b,
                 float* __restrict__ ybuf) {     // [R,128]
  __shared__ float xr[48];
  const int c = threadIdx.x;
  const long long m = blockIdx.x;
  float Ac[16];
#pragma unroll
  for (int s = 0; s < 16; ++s) Ac[s] = -__expf(A_log[c * 16 + s]);
  const float w0 = dt_w[c * 4 + 0], w1 = dt_w[c * 4 + 1];
  const float w2 = dt_w[c * 4 + 2], w3 = dt_w[c * 4 + 3];
  const float db = dt_b[c], Dc = D[c];
  float h[16];
#pragma unroll
  for (int s = 0; s < 16; ++s) h[s] = 0.f;

  for (int t = 0; t < 36; ++t) {
    long long row = m * 36 + t;
    if (c < 48) xr[c] = xdbl[row * 48 + c];
    __syncthreads();
    float dt = xr[0] * w0 + xr[1] * w1 + xr[2] * w2 + xr[3] * w3 + db;
    float delta = (dt > 20.f) ? dt : log1pf(__expf(dt));
    float u  = xa[row * 128 + c];
    float du = delta * u;
    float y = 0.f;
#pragma unroll
    for (int s = 0; s < 16; ++s) {
      float dA = __expf(delta * Ac[s]);
      h[s] = dA * h[s] + du * xr[4 + s];
      y += h[s] * xr[20 + s];
    }
    float zg = xz[row * 256 + 128 + c];
    ybuf[row * 128 + c] = (y + u * Dc) * (zg * sigmoidf_(zg));
    __syncthreads();
  }
}

// Output head: o1 = relu(einsum over t) ; out = einsum over d
__global__ __launch_bounds__(64)
void head_kernel(const float* __restrict__ om, const float* __restrict__ w1,
                 const float* __restrict__ b1, const float* __restrict__ w2,
                 const float* __restrict__ b2, float* __restrict__ out) {
  __shared__ float red[64];
  const long long bn = blockIdx.x;
  const int d = threadIdx.x;
  float v[36];
#pragma unroll
  for (int t = 0; t < 36; ++t) v[t] = om[(bn * 36 + t) * 64 + d];
  const float w2d = w2[d];
  for (int o = 0; o < 12; ++o) {
    float s = b1[o];
#pragma unroll
    for (int t = 0; t < 36; ++t) s += v[t] * w1[o * 36 + t];
    red[d] = fmaxf(s, 0.f) * w2d;
    __syncthreads();
    for (int off = 32; off > 0; off >>= 1) {
      if (d < off) red[d] += red[d + off];
      __syncthreads();
    }
    if (d == 0) out[bn * 12 + o] = red[0] + b2[0];
    __syncthreads();
  }
}

// ---------------------------------------------------------------------------
extern "C" void kernel_launch(void* const* d_in, const int* in_sizes, int n_in,
                              void* d_out, int out_size, void* d_ws, size_t ws_size,
                              hipStream_t stream) {
  (void)in_sizes; (void)n_in; (void)out_size; (void)ws_size;
  auto F = [&](int i) { return (const float*)d_in[i]; };

  const float* x = F(0);                 // [8,1,512,36] == [4096,36] (din=1)
  // d_in[1] = time_features (unused by reference)
  struct GruP { const float *Wih, *Whh, *bih, *bhh; };
  GruP gs{F(2), F(3), F(4), F(5)};
  GruP gl{F(6), F(7), F(8), F(9)};
  const float* nv1 = F(10);
  const float* nv2 = F(11);
  const float* gcn_w = F(12);
  const float* gcn_b = F(13);
  const float* n1g = F(14); const float* n1b = F(15);
  const float* n2g = F(16); const float* n2b = F(17);
  struct MambaP { const float *in_proj, *conv_w, *conv_b, *x_proj,
                              *dt_w, *dt_b, *A_log, *D, *out_proj; };
  MambaP mf{F(18), F(19), F(20), F(21), F(22), F(23), F(24), F(25), F(26)};
  MambaP mb{F(27), F(28), F(29), F(30), F(31), F(32), F(33), F(34), F(35)};
  const float* c1w = F(36); const float* c1b = F(37);
  const float* c2w = F(38); const float* c2b = F(39);
  float* outp = (float*)d_out;

  const long long R = 8LL * 512 * 36;    // 147456 rows (b,n,t)
  float* W = (float*)d_ws;
  float* seq  = W;                       // [R,64]  persistent
  float* acc  = seq  + R * 64;           // [R,64]  fwd mamba result
  float* ybwd = acc  + R * 64;           // [R,64]  bwd mamba result
  float* Sb   = ybwd + R * 64;           // overlay scratch base
  // phase 2 (GCN)
  float* hbuf = Sb;                      // [R,64] == [B,512,2304] per batch
  float* adjT = hbuf + R * 64;           // [512,512]
  float* x1   = adjT + 512LL * 512;      // [R,64]
  float* x2   = x1 + R * 64;             // [R,64]
  float* cat  = x2 + R * 64;             // [R,192]
  float* gres = cat + R * 192;           // [R,64]
  // phase 3 (Mamba) overlays phase 2
  float* seqR = Sb;                      // [R,64]
  float* xzb  = seqR + R * 64;           // [R,256]
  float* xab  = xzb + R * 256;           // [R,128]
  float* xdb  = xab + R * 128;           // [R,48]
  float* ysc  = xdb + R * 48;            // [R,128]
  // phase 4
  float* om   = Sb;                      // [R,64]

  // ---- GRU encoders (week / day with long weights, short with short weights)
  gru_kernel<<<256, 32, 0, stream>>>(x, gl.Wih, gl.Whh, gl.bih, gl.bhh, hbuf, 0, 12);
  gru_kernel<<<256, 32, 0, stream>>>(x, gl.Wih, gl.Whh, gl.bih, gl.bhh, hbuf, 12, 12);
  gru_kernel<<<256, 32, 0, stream>>>(x, gs.Wih, gs.Whh, gs.bih, gs.bhh, hbuf, 24, 12);

  // ---- adaptive adjacency + 2-hop graph conv
  adj_kernel<<<512, 256, 0, stream>>>(nv1, nv2, adjT);
  dim3 gg(512 / 64, 2304 / 64, 8);
  gemm_wmma<<<gg, 128, 0, stream>>>(adjT, hbuf, nullptr, nullptr, x1,
      512, 512, 2304, 2304, /*b_layout*/1, 2304, 0, 512LL * 2304, 512LL * 2304);
  gemm_wmma<<<gg, 128, 0, stream>>>(adjT, x1, nullptr, nullptr, x2,
      512, 512, 2304, 2304, 1, 2304, 0, 512LL * 2304, 512LL * 2304);
  concat_kernel<<<(unsigned)((R * 192 + 255) / 256), 256, 0, stream>>>(
      hbuf, x1, x2, cat, R * 192);
  dim3 gp((unsigned)(R / 64), 1, 1);
  gemm_wmma<<<gp, 128, 0, stream>>>(cat, gcn_w, gcn_b, /*res=*/hbuf, gres,
      192, 192, 192, 64, 0, 64, 0, 0, 0);
  ln_kernel<<<(unsigned)(R / 8), 256, 0, stream>>>(gres, nullptr, nullptr,
      n1g, n1b, seq, (int)R, 0);

  // ---- Mamba block (shared pipeline for both directions)
  auto run_mamba = [&](const float* sIn, const MambaP& p, float* yout) {
    dim3 gi((unsigned)(R / 64), 256 / 64, 1);
    gemm_wmma<<<gi, 128, 0, stream>>>(sIn, p.in_proj, nullptr, nullptr, xzb,
        64, 64, 64, 256, 0, 256, 0, 0, 0);
    dwconv_silu_kernel<<<(unsigned)((R * 128 + 255) / 256), 256, 0, stream>>>(
        xzb, p.conv_w, p.conv_b, xab, R * 128);
    dim3 gx((unsigned)(R / 64), 1, 1);
    gemm_wmma<<<gx, 128, 0, stream>>>(xab, p.x_proj, nullptr, nullptr, xdb,
        128, 128, 128, 48, 0, /*Nreal=*/36, 0, 0, 0);
    scan_kernel<<<4096, 128, 0, stream>>>(xdb, xab, xzb, p.A_log, p.D,
        p.dt_w, p.dt_b, ysc);
    gemm_wmma<<<gx, 128, 0, stream>>>(ysc, p.out_proj, nullptr, nullptr, yout,
        128, 128, 128, 64, 0, 64, 0, 0, 0);
  };
  run_mamba(seq, mf, acc);
  flip_kernel<<<(unsigned)((R * 64 + 255) / 256), 256, 0, stream>>>(seq, seqR, R * 64);
  run_mamba(seqR, mb, ybwd);

  // ---- combine fwd + flipped bwd + residual, LayerNorm2
  ln_kernel<<<(unsigned)(R / 8), 256, 0, stream>>>(acc, ybwd, seq,
      n2g, n2b, om, (int)R, /*flip2=*/1);

  // ---- output head
  head_kernel<<<4096, 64, 0, stream>>>(om, c1w, c1b, c2w, c2b, outp);
}
